// AttentionCritic_34754875359696
// MI455X (gfx1250) — matrix operand order repaired and checked by reading, not compiled, not run on hardware
//
#include <hip/hip_runtime.h>

// ---------------- problem constants ----------------
#define AG    8
#define BB    8192
#define SS    128
#define ADIMC 16
#define HHC   512
#define NHEAD 4
#define DDC   128
#define EPSLN 1e-5f

// ---------------- types ----------------
typedef __attribute__((ext_vector_type(16))) _Float16 v16h;
typedef __attribute__((ext_vector_type(8)))  float    v8f;

union H16 { uint4 q[2]; _Float16 h[16]; unsigned u[8]; };
union H4  { uint2 u;    _Float16 h[4]; };

// ---------------- WMMA fragment loaders (ISA 7.12.2 layouts) ----------------
// A (16x32 f16): lanes 0-15 hold M=0..15 (lanes 16-31 repeat M with K+8 offset);
// v0..3 = K pairs {kb+2i, kb+2i+1}, v4..7 = {16+kb+2i, ...}, kb=(lane>>4)*8.
// LDS tile row-major, stride 40 halfs (pad vs bank conflicts).
__device__ __forceinline__ v16h load_frag_a(const _Float16* lds, int rowBase, int lane) {
    int r  = rowBase + (lane & 15);
    int kb = (lane >> 4) * 8;
    const _Float16* p = lds + r * 40;
    H16 f;
#pragma unroll
    for (int i = 0; i < 4; ++i) f.u[i]     = *(const unsigned*)(p + kb + 2 * i);
#pragma unroll
    for (int i = 0; i < 4; ++i) f.u[4 + i] = *(const unsigned*)(p + 16 + kb + 2 * i);
    v16h v;
#pragma unroll
    for (int i = 0; i < 16; ++i) v[i] = f.h[i];
    return v;
}

// B (32x16 f16): lane holds column N=lane&15; same K order as A:
// v0..3 = K pairs {kb2+2i, kb2+2i+1}, v4..7 = {16+kb2+2i, ...}, kb2=(lane>>4)*8.
// LDS tile N-major: ldsB[col*40 + k] so pairs are contiguous (ds_load_b128 friendly).
__device__ __forceinline__ v16h load_frag_b(const _Float16* lds, int colBase, int lane) {
    int c   = colBase + (lane & 15);
    int kb2 = (lane >> 4) * 8;
    const _Float16* p = lds + c * 40 + kb2;
    H16 f;
#pragma unroll
    for (int i = 0; i < 8; ++i) f.u[i] = *(const unsigned*)(p + (i >> 2) * 16 + 2 * (i & 3));
    v16h v;
#pragma unroll
    for (int i = 0; i < 16; ++i) v[i] = f.h[i];
    return v;
}

// ---------------- kernel 0: LayerNorm prep + argmax (wave per row) ----------------
__global__ void __launch_bounds__(256)
ln_prep_kernel(const float* __restrict__ states, const float* __restrict__ actions,
               _Float16* __restrict__ lnS, _Float16* __restrict__ lnSA,
               int* __restrict__ idxb)
{
    int t = threadIdx.x, lane = t & 31, wid = t >> 5;
    size_t r = (size_t)blockIdx.x * 8 + wid;

    float4 x = *(const float4*)(states + r * SS + lane * 4);
    float s1 = x.x + x.y + x.z + x.w;
    float s2 = x.x * x.x + x.y * x.y + x.z * x.z + x.w * x.w;
#pragma unroll
    for (int m = 16; m; m >>= 1) { s1 += __shfl_xor(s1, m, 32); s2 += __shfl_xor(s2, m, 32); }
    float mean = s1 * (1.f / 128.f);
    float var  = s2 * (1.f / 128.f) - mean * mean;
    float rs   = rsqrtf(var + EPSLN);
    { H4 pk;
      pk.h[0] = (_Float16)((x.x - mean) * rs); pk.h[1] = (_Float16)((x.y - mean) * rs);
      pk.h[2] = (_Float16)((x.z - mean) * rs); pk.h[3] = (_Float16)((x.w - mean) * rs);
      *(uint2*)(lnS + r * SS + lane * 4) = pk.u; }

    float av = (lane < ADIMC) ? actions[r * ADIMC + lane] : 0.f;
    float a1 = av, a2 = av * av;
#pragma unroll
    for (int m = 16; m; m >>= 1) { a1 += __shfl_xor(a1, m, 32); a2 += __shfl_xor(a2, m, 32); }
    float mean2 = (s1 + a1) * (1.f / 144.f);
    float var2  = (s2 + a2) * (1.f / 144.f) - mean2 * mean2;
    float rs2   = rsqrtf(var2 + EPSLN);
    { H4 pk;
      pk.h[0] = (_Float16)((x.x - mean2) * rs2); pk.h[1] = (_Float16)((x.y - mean2) * rs2);
      pk.h[2] = (_Float16)((x.z - mean2) * rs2); pk.h[3] = (_Float16)((x.w - mean2) * rs2);
      *(uint2*)(lnSA + r * 160 + lane * 4) = pk.u; }
    if (lane < ADIMC) {
        lnSA[r * 160 + 128 + lane] = (_Float16)((av - mean2) * rs2);
        lnSA[r * 160 + 144 + lane] = (_Float16)0.f;
    }

    float mv = (lane < ADIMC) ? av : -3.4e38f;
    int   mi = (lane < ADIMC) ? lane : 1 << 20;
#pragma unroll
    for (int m = 16; m; m >>= 1) {
        float ov = __shfl_xor(mv, m, 32);
        int   oi = __shfl_xor(mi, m, 32);
        if (ov > mv || (ov == mv && oi < mi)) { mv = ov; mi = oi; }
    }
    if (lane == 0) idxb[r] = mi;
}

// ---------------- weight convert + transpose ----------------
// src: f32 [Z][Ksrc][N] (row-major)  ->  dst: f16 [Z][N][Kp] (N-major, K zero-padded)
__global__ void __launch_bounds__(256)
cvt_t_kernel(const float* __restrict__ src, _Float16* __restrict__ dst,
             int Ksrc, int Kp, int N, int total)
{
    int i = blockIdx.x * 256 + threadIdx.x;
    if (i >= total) return;
    int k = i % Kp;
    int n = (i / Kp) % N;
    int z = i / (Kp * N);
    dst[i] = (k < Ksrc) ? (_Float16)src[((size_t)z * Ksrc + k) * N + n] : (_Float16)0.f;
}

// ---------------- generic batched WMMA GEMM ----------------
// C[z][8192, N] = act( LN?( concat(A0,A1)[z] ) x B[z] + bias[z] ),  B is N-major [N][K].
__global__ void __launch_bounds__(256)
wmma_gemm_kernel(const _Float16* __restrict__ A0, const _Float16* __restrict__ A1,
                 int seamK, const float* __restrict__ stats,
                 const _Float16* __restrict__ Bw, const float* __restrict__ bias,
                 _Float16* __restrict__ C,
                 int N, int K, int lda0, int lda1, int ldbT, int ldc,
                 int zmod, long long sA, long long sStats,
                 long long sB1, long long sB2,
                 long long sBias1, long long sBias2, long long sC, int act)
{
    __shared__ _Float16 ldsA[128 * 40];
    __shared__ _Float16 ldsB[128 * 40];

    const int t    = threadIdx.x;
    const int lane = t & 31;
    const int wid  = t >> 5;
    const int wm   = wid >> 1;
    const int wn   = wid & 1;

    const long long z  = blockIdx.z;
    const long long az = z % zmod, bz = z / zmod;
    A0 += az * sA;
    if (A1)    A1    += az * sA;
    if (stats) stats += az * sStats;
    Bw += az * sB1 + bz * sB2;
    if (bias) bias += az * sBias1 + bz * sBias2;
    C += z * sC;

    const int rbase = blockIdx.x * 128;
    const int cbase = blockIdx.y * 128;

    v8f vzero = {0.f, 0.f, 0.f, 0.f, 0.f, 0.f, 0.f, 0.f};
    v8f acc[8];
#pragma unroll
    for (int i = 0; i < 8; ++i) acc[i] = vzero;

    const int arow = t >> 1;            // 0..127 (row for A, col for B)
    const int aseg = (t & 1) * 16;      // 0 or 16

    float mean = 0.f, rstd = 1.f;
    if (stats) {
        mean = stats[(size_t)(rbase + arow) * 2];
        rstd = stats[(size_t)(rbase + arow) * 2 + 1];
    }

    for (int kt = 0; kt < K; kt += 32) {
        {   // stage A (128x32), contiguous
            int kk = kt + aseg;
            const _Float16* src = (kk < seamK)
                ? (A0 + (size_t)(rbase + arow) * lda0 + kk)
                : (A1 + (size_t)(rbase + arow) * lda1 + (kk - seamK));
            if (kt + 32 < K) __builtin_prefetch(src + 32, 0, 1);
            H16 v;
            v.q[0] = *(const uint4*)src;
            v.q[1] = *(const uint4*)(src + 8);
            if (stats) {
#pragma unroll
                for (int j = 0; j < 16; ++j)
                    v.h[j] = (_Float16)(((float)v.h[j] - mean) * rstd);
            }
            *(uint4*)&ldsA[arow * 40 + aseg]     = v.q[0];
            *(uint4*)&ldsA[arow * 40 + aseg + 8] = v.q[1];
        }
        {   // stage B (cols cbase..+127, k kt..+31) from N-major weights, contiguous
            const _Float16* src = Bw + (size_t)(cbase + arow) * ldbT + kt + aseg;
            if (kt + 32 < K) __builtin_prefetch(src + 32, 0, 1);
            uint4 u0 = *(const uint4*)src;
            uint4 u1 = *(const uint4*)(src + 8);
            *(uint4*)&ldsB[arow * 40 + aseg]     = u0;
            *(uint4*)&ldsB[arow * 40 + aseg + 8] = u1;
        }
        __syncthreads();

        v16h af[2], bf[4];
#pragma unroll
        for (int mi = 0; mi < 2; ++mi) af[mi] = load_frag_a(ldsA, wm * 32 + mi * 16, lane);
#pragma unroll
        for (int ni = 0; ni < 4; ++ni) bf[ni] = load_frag_b(ldsB, wn * 64 + ni * 16, lane);
#pragma unroll
        for (int mi = 0; mi < 2; ++mi)
#pragma unroll
            for (int ni = 0; ni < 4; ++ni)
                acc[mi * 4 + ni] = __builtin_amdgcn_wmma_f32_16x16x32_f16(
                    false, af[mi], false, bf[ni], (short)0, acc[mi * 4 + ni], false, false);
        __syncthreads();
    }

#pragma unroll
    for (int mi = 0; mi < 2; ++mi) {
#pragma unroll
        for (int ni = 0; ni < 4; ++ni) {
            int col  = cbase + wn * 64 + ni * 16 + (lane & 15);
            float bv = bias ? bias[col] : 0.f;
#pragma unroll
            for (int i = 0; i < 8; ++i) {
                int row = rbase + wm * 32 + mi * 16 + ((lane < 16) ? i : i + 8);
                float v = acc[mi * 4 + ni][i] + bv;
                if (act) v = (v > 0.f) ? v : 0.01f * v;
                C[(size_t)row * ldc + col] = (_Float16)v;
            }
        }
    }
}

// ---------------- attention: wave per (head, b) ----------------
__global__ void __launch_bounds__(256)
attn_kernel(const _Float16* __restrict__ keys, const _Float16* __restrict__ vals,
            const _Float16* __restrict__ sel, _Float16* __restrict__ outv)
{
    __shared__ _Float16 shS[8][AG][DDC];
    __shared__ _Float16 shK[8][AG][DDC];
    __shared__ _Float16 shV[8][AG][DDC];
    __shared__ float    shW[8][64];

    int t = threadIdx.x, lane = t & 31, wid = t >> 5;
    long long w = (long long)blockIdx.x * 8 + wid;
    int k = (int)(w / BB);
    int b = (int)(w % BB);

#pragma unroll
    for (int a = 0; a < AG; ++a) {
        size_t base = (((size_t)k * AG + a) * BB + b) * DDC + lane * 4;
        *(uint2*)&shS[wid][a][lane * 4] = *(const uint2*)(sel  + base);
        *(uint2*)&shK[wid][a][lane * 4] = *(const uint2*)(keys + base);
        *(uint2*)&shV[wid][a][lane * 4] = *(const uint2*)(vals + base);
    }
    __syncthreads();

    const float scale = 0.0883883476483184f; // 1/sqrt(128)
#pragma unroll
    for (int tt = 0; tt < 2; ++tt) {
        int p = lane + 32 * tt;
        int i = p >> 3, j = p & 7;
        float acc = 0.f;
        for (int d = 0; d < DDC; ++d)
            acc += (float)shS[wid][i][d] * (float)shK[wid][j][d];
        shW[wid][p] = acc * scale;
    }
    __syncthreads();

    if (lane < AG) {
        int i = lane;
        float m = -3.4e38f;
#pragma unroll
        for (int j = 0; j < AG; ++j) if (j != i) m = fmaxf(m, shW[wid][i * 8 + j]);
        float e[AG], s = 0.f;
#pragma unroll
        for (int j = 0; j < AG; ++j) {
            e[j] = (j == i) ? 0.f : __expf(shW[wid][i * 8 + j] - m);
            s += e[j];
        }
        float inv = 1.f / s;
#pragma unroll
        for (int j = 0; j < AG; ++j) shW[wid][i * 8 + j] = e[j] * inv;
    }
    __syncthreads();

    int d0 = lane * 4;
#pragma unroll
    for (int i = 0; i < AG; ++i) {
        float o0 = 0.f, o1 = 0.f, o2 = 0.f, o3 = 0.f;
#pragma unroll
        for (int j = 0; j < AG; ++j) {
            float ww = shW[wid][i * 8 + j];
            o0 += ww * (float)shV[wid][j][d0 + 0];
            o1 += ww * (float)shV[wid][j][d0 + 1];
            o2 += ww * (float)shV[wid][j][d0 + 2];
            o3 += ww * (float)shV[wid][j][d0 + 3];
        }
        H4 pk;
        pk.h[0] = (_Float16)o0; pk.h[1] = (_Float16)o1;
        pk.h[2] = (_Float16)o2; pk.h[3] = (_Float16)o3;
        *(uint2*)(outv + (((size_t)i * BB + b) * HHC + k * DDC + d0)) = pk.u;
    }
}

// ---------------- cin LayerNorm stats ----------------
__global__ void __launch_bounds__(256)
cin_stats_kernel(const _Float16* __restrict__ s_enc, const _Float16* __restrict__ values,
                 float* __restrict__ stats)
{
    int t = threadIdx.x, lane = t & 31, wid = t >> 5;
    size_t r = (size_t)blockIdx.x * 8 + wid;
    float s1 = 0.f, s2 = 0.f;
#pragma unroll
    for (int c = 0; c < 4; ++c) {
        H4 v; v.u = *(const uint2*)(s_enc + r * HHC + c * 128 + lane * 4);
#pragma unroll
        for (int j = 0; j < 4; ++j) { float x = (float)v.h[j]; s1 += x; s2 += x * x; }
    }
#pragma unroll
    for (int c = 0; c < 4; ++c) {
        H4 v; v.u = *(const uint2*)(values + r * HHC + c * 128 + lane * 4);
#pragma unroll
        for (int j = 0; j < 4; ++j) { float x = (float)v.h[j]; s1 += x; s2 += x * x; }
    }
#pragma unroll
    for (int m = 16; m; m >>= 1) { s1 += __shfl_xor(s1, m, 32); s2 += __shfl_xor(s2, m, 32); }
    if (lane == 0) {
        float mean = s1 * (1.f / 1024.f);
        float var  = s2 * (1.f / 1024.f) - mean * mean;
        stats[r * 2]     = mean;
        stats[r * 2 + 1] = rsqrtf(var + EPSLN);
    }
}

// ---------------- final critic layer + gather ----------------
// all_qs = h1[128,512] x Wc2T[16][512]^T + bc2 ; out[row] = all_qs[row][idx[row]]
__global__ void __launch_bounds__(256)
critic_out_kernel(const _Float16* __restrict__ h1, const _Float16* __restrict__ w2T,
                  const float* __restrict__ b2, const int* __restrict__ idxb,
                  float* __restrict__ out)
{
    __shared__ _Float16 ldsA[128 * 40];
    __shared__ _Float16 ldsB[16 * 40];
    __shared__ float    shQ[8][16][16];

    int t = threadIdx.x, lane = t & 31, wid = t >> 5;
    int a = blockIdx.y;
    int rbase = blockIdx.x * 128;
    const _Float16* A = h1  + (size_t)a * BB * HHC;
    const _Float16* W = w2T + (size_t)a * ADIMC * HHC;   // [16][512] N-major

    v8f acc = {0.f, 0.f, 0.f, 0.f, 0.f, 0.f, 0.f, 0.f};
    int arow = t >> 1, aseg = (t & 1) * 16;

    for (int kt = 0; kt < HHC; kt += 32) {
        const _Float16* src = A + (size_t)(rbase + arow) * HHC + kt + aseg;
        uint4 u0 = *(const uint4*)src;
        uint4 u1 = *(const uint4*)(src + 8);
        *(uint4*)&ldsA[arow * 40 + aseg]     = u0;
        *(uint4*)&ldsA[arow * 40 + aseg + 8] = u1;
        if (t < 32) {   // 16 cols x 32 k, contiguous from N-major weights
            int n = t >> 1, ks = (t & 1) * 16;
            const _Float16* wsrc = W + (size_t)n * HHC + kt + ks;
            uint4 w0 = *(const uint4*)wsrc;
            uint4 w1 = *(const uint4*)(wsrc + 8);
            *(uint4*)&ldsB[n * 40 + ks]     = w0;
            *(uint4*)&ldsB[n * 40 + ks + 8] = w1;
        }
        __syncthreads();
        v16h af = load_frag_a(ldsA, wid * 16, lane);
        v16h bf = load_frag_b(ldsB, 0, lane);
        acc = __builtin_amdgcn_wmma_f32_16x16x32_f16(false, af, false, bf,
                                                     (short)0, acc, false, false);
        __syncthreads();
    }

    int col = lane & 15;
    float bv = b2[a * ADIMC + col];
#pragma unroll
    for (int i = 0; i < 8; ++i)
        shQ[wid][(lane < 16) ? i : i + 8][col] = acc[i] + bv;
    __syncthreads();
    if (lane < 16) {
        int row = rbase + wid * 16 + lane;
        int id  = idxb[(size_t)a * BB + row];
        out[(size_t)a * BB + row] = shQ[wid][lane][id];
    }
}

// ---------------- host launcher ----------------
extern "C" void kernel_launch(void* const* d_in, const int* in_sizes, int n_in,
                              void* d_out, int out_size, void* d_ws, size_t ws_size,
                              hipStream_t stream)
{
    (void)in_sizes; (void)n_in; (void)out_size; (void)ws_size;

    const float* states  = (const float*)d_in[0];
    const float* actions = (const float*)d_in[1];
    const float* Ws_s    = (const float*)d_in[2];
    const float* bs_s    = (const float*)d_in[3];
    const float* Ws_sa   = (const float*)d_in[4];
    const float* bs_sa   = (const float*)d_in[5];
    const float* Wk      = (const float*)d_in[6];
    const float* Wv      = (const float*)d_in[7];
    const float* Wsel    = (const float*)d_in[8];
    const float* bsel    = (const float*)d_in[9];
    const float* Wc1     = (const float*)d_in[10];
    const float* bc1     = (const float*)d_in[11];
    const float* Wc2     = (const float*)d_in[12];
    const float* bc2     = (const float*)d_in[13];
    float* out = (float*)d_out;

    char* ws = (char*)d_ws;
    size_t off = 0;
    auto alloc = [&](size_t bytes) -> char* {
        off = (off + 255) & ~(size_t)255;
        char* p = ws + off;
        off += bytes;
        return p;
    };
    const size_t nEnc = (size_t)AG * BB * HHC;

    // transposed (N-major) f16 weights
    _Float16* wWs_s  = (_Float16*)alloc((size_t)AG * HHC * SS * 2);        // [A][512][128]
    _Float16* wWs_sa = (_Float16*)alloc((size_t)AG * HHC * 160 * 2);       // [A][512][160]
    _Float16* wWk    = (_Float16*)alloc((size_t)NHEAD * DDC * HHC * 2);    // [4][128][512]
    _Float16* wWv    = (_Float16*)alloc((size_t)NHEAD * DDC * HHC * 2);
    _Float16* wWsel  = (_Float16*)alloc((size_t)NHEAD * DDC * HHC * 2);
    _Float16* wWc1   = (_Float16*)alloc((size_t)AG * HHC * 1024 * 2);      // [A][512][1024]
    _Float16* wWc2   = (_Float16*)alloc((size_t)AG * ADIMC * HHC * 2);     // [A][16][512]
    int*      idxb   = (int*)alloc((size_t)AG * BB * 4);
    float*    stats  = (float*)alloc((size_t)AG * BB * 2 * 4);
    _Float16* ln_s   = (_Float16*)alloc((size_t)AG * BB * SS * 2);
    _Float16* ln_sa  = (_Float16*)alloc((size_t)AG * BB * 160 * 2);
    _Float16* s_enc  = (_Float16*)alloc(nEnc * 2);
    _Float16* bufV   = (_Float16*)alloc(nEnc * 2);   // sa_enc -> attended values
    _Float16* bufKH  = (_Float16*)alloc(nEnc * 2);   // keys   -> h1
    _Float16* valsb  = (_Float16*)alloc(nEnc * 2);
    _Float16* selb   = (_Float16*)alloc(nEnc * 2);

    const dim3 blk(256);
    const long long LB = BB;

    ln_prep_kernel<<<AG * BB / 8, blk, 0, stream>>>(states, actions, ln_s, ln_sa, idxb);

    auto cvtT = [&](const float* s, _Float16* d, int Z, int Ksrc, int Kp, int N) {
        int total = Z * N * Kp;
        cvt_t_kernel<<<(total + 255) / 256, blk, 0, stream>>>(s, d, Ksrc, Kp, N, total);
    };
    cvtT(Ws_s,  wWs_s,  AG, SS, SS, HHC);
    cvtT(Ws_sa, wWs_sa, AG, 144, 160, HHC);
    cvtT(Wk,    wWk,    NHEAD, HHC, HHC, DDC);
    cvtT(Wv,    wWv,    NHEAD, HHC, HHC, DDC);
    cvtT(Wsel,  wWsel,  NHEAD, HHC, HHC, DDC);
    cvtT(Wc1,   wWc1,   AG, 1024, 1024, HHC);
    cvtT(Wc2,   wWc2,   AG, HHC, HHC, ADIMC);

    // s_enc = lrelu(LN(states) @ Ws_s + bs_s)
    wmma_gemm_kernel<<<dim3(BB / 128, HHC / 128, AG), blk, 0, stream>>>(
        ln_s, nullptr, SS, nullptr, wWs_s, bs_s, s_enc,
        HHC, SS, SS, 0, SS, HHC,
        AG, LB * SS, 0, (long long)HHC * SS, 0, HHC, 0, LB * HHC, 1);

    // sa_enc = lrelu(LN(sa) @ Ws_sa + bs_sa)
    wmma_gemm_kernel<<<dim3(BB / 128, HHC / 128, AG), blk, 0, stream>>>(
        ln_sa, nullptr, 160, nullptr, wWs_sa, bs_sa, bufV,
        HHC, 160, 160, 0, 160, HHC,
        AG, LB * 160, 0, (long long)HHC * 160, 0, HHC, 0, LB * HHC, 1);

    // keys / vals / sel : z = head*AG + agent
    wmma_gemm_kernel<<<dim3(BB / 128, 1, NHEAD * AG), blk, 0, stream>>>(
        bufV, nullptr, HHC, nullptr, wWk, nullptr, bufKH,
        DDC, HHC, HHC, 0, HHC, DDC,
        AG, LB * HHC, 0, 0, (long long)DDC * HHC, 0, 0, LB * DDC, 0);
    wmma_gemm_kernel<<<dim3(BB / 128, 1, NHEAD * AG), blk, 0, stream>>>(
        bufV, nullptr, HHC, nullptr, wWv, nullptr, valsb,
        DDC, HHC, HHC, 0, HHC, DDC,
        AG, LB * HHC, 0, 0, (long long)DDC * HHC, 0, 0, LB * DDC, 0);
    wmma_gemm_kernel<<<dim3(BB / 128, 1, NHEAD * AG), blk, 0, stream>>>(
        bufV, nullptr, HHC, nullptr, wWsel, bsel, selb,
        DDC, HHC, HHC, 0, HHC, DDC,
        AG, LB * HHC, 0, 0, (long long)DDC * HHC, 0, DDC, LB * DDC, 1);

    attn_kernel<<<NHEAD * BB / 8, blk, 0, stream>>>(bufKH, valsb, selb, bufV);

    cin_stats_kernel<<<AG * BB / 8, blk, 0, stream>>>(s_enc, bufV, stats);

    // h1 = lrelu( LN(concat(s_enc, values)) @ Wc1 + bc1 )
    wmma_gemm_kernel<<<dim3(BB / 128, HHC / 128, AG), blk, 0, stream>>>(
        s_enc, bufV, HHC, stats, wWc1, bc1, bufKH,
        HHC, 2 * HHC, HHC, HHC, 2 * HHC, HHC,
        AG, LB * HHC, LB * 2, (long long)HHC * 2 * HHC, 0, HHC, 0, LB * HHC, 1);

    critic_out_kernel<<<dim3(BB / 128, AG), blk, 0, stream>>>(bufKH, wWc2, bc2, idxb, out);
}
